// SpatialLSTM_67697274519651
// MI455X (gfx1250) — matrix operand order repaired and checked by reading, not compiled, not run on hardware
//
#include <hip/hip_runtime.h>
#include <hip/hip_bf16.h>

typedef __attribute__((ext_vector_type(16))) __bf16 v16bf;
typedef __attribute__((ext_vector_type(8)))  __bf16 v8bf;
typedef __attribute__((ext_vector_type(8)))  float  v8f;

#define B_   64
#define T_   2048
#define D_   128
#define H_   256
#define G4_  (4 * H_)
#define NWG  16
#define PADW (D_ + 8)
#define PADU (H_ + 8)

#define WS_XBF_OFF  131072          // Xbf16 offset in ws (barrier + hbuf live below)
#define WS_XBF_SZ   ((size_t)B_ * T_ * D_ * 2)

__device__ __forceinline__ float sigmoidf_(float x) {
    return 1.0f / (1.0f + __expf(-x));
}

// Reset grid-barrier counter and seed h-buffer[0] with bf16(h0). Must run every
// launch (graph replay reuses ws without re-poisoning).
__global__ __launch_bounds__(256) void lstm_init_kernel(const float* __restrict__ h0,
                                                        __bf16* __restrict__ hbuf,
                                                        unsigned* __restrict__ bar) {
    int i = blockIdx.x * blockDim.x + threadIdx.x;
    if (i == 0) *bar = 0u;
    if (i < B_ * H_) hbuf[i] = (__bf16)h0[i];
}

// One-time X fp32 -> bf16 pre-conversion (removes per-step v_cvt from the scan).
__global__ __launch_bounds__(256) void x_to_bf16_kernel(const float* __restrict__ X,
                                                        __bf16* __restrict__ Xbf) {
    size_t i = ((size_t)blockIdx.x * blockDim.x + threadIdx.x) * 8;
    if (i >= (size_t)B_ * T_ * D_) return;
    float4 f0 = *(const float4*)(X + i);
    float4 f1 = *(const float4*)(X + i + 4);
    v8bf o;
    o[0] = (__bf16)f0.x; o[1] = (__bf16)f0.y; o[2] = (__bf16)f0.z; o[3] = (__bf16)f0.w;
    o[4] = (__bf16)f1.x; o[5] = (__bf16)f1.y; o[6] = (__bf16)f1.z; o[7] = (__bf16)f1.w;
    *(v8bf*)(Xbf + i) = o;
}

template <bool PRECONV>
__global__ __launch_bounds__(128) void lstm_persistent_kernel(
    const float* __restrict__ X, const __bf16* __restrict__ Xbf,
    const float* __restrict__ W, const float* __restrict__ U,
    const float* __restrict__ bias, const float* __restrict__ c0,
    float* __restrict__ out, __bf16* __restrict__ hbuf,
    unsigned* __restrict__ bar)
{
    // LDS-resident weight slices, bf16, column-major per gate column (padded rows
    // so the 16 lanes of a b128 read land on distinct bank groups).
    __shared__ __align__(16) __bf16 ldsW[4][16][PADW];   // [gate][col][k] k<128
    __shared__ __align__(16) __bf16 ldsU[4][16][PADU];   // [gate][col][k] k<256

    const int wg   = blockIdx.x;          // N-slice: hidden cols [16*wg, 16*wg+16)
    const int tid  = threadIdx.x;
    const int wave = tid >> 5;            // M-tile: batch rows [16*wave, 16*wave+16)
    const int lane = tid & 31;
    const int col  = lane & 15;
    const int hi   = lane >> 4;
    const int b0   = wave * 16;
    const int arow = b0 + col;            // A-matrix row this lane loads
    const int jc   = wg * 16 + col;       // hidden column this lane owns (C layout)

    // ---- one-time staging of W and U slices into LDS (bf16) ----
    for (int e = tid; e < 4 * 16 * D_; e += 128) {
        int g = e / (16 * D_); int r = e % (16 * D_); int c = r / D_; int k = r % D_;
        ldsW[g][c][k] = (__bf16)W[(size_t)k * G4_ + g * H_ + wg * 16 + c];
    }
    for (int e = tid; e < 4 * 16 * H_; e += 128) {
        int g = e / (16 * H_); int r = e % (16 * H_); int c = r / H_; int k = r % H_;
        ldsU[g][c][k] = (__bf16)U[(size_t)k * G4_ + g * H_ + wg * 16 + c];
    }
    __syncthreads();

    // Per-lane bias for each gate (same for every row in the C tile).
    float biasv[4];
#pragma unroll
    for (int g = 0; g < 4; ++g) biasv[g] = bias[g * H_ + jc];

    // Cell state lives in registers in WMMA-C layout: creg[r] = c[b0 + r + 8*hi][jc]
    v8f creg;
#pragma unroll
    for (int r = 0; r < 8; ++r) creg[r] = c0[(size_t)(b0 + r + hi * 8) * H_ + jc];
    v8f hreg = {};

    const float*  xrow0f = X   + ((size_t)arow * T_) * D_;
    const __bf16* xrow0b = Xbf + ((size_t)arow * T_) * D_;

    for (int t = 0; t < T_; ++t) {
        v8f acc[4];
#pragma unroll
        for (int g = 0; g < 4; ++g)
#pragma unroll
            for (int r = 0; r < 8; ++r) acc[g][r] = biasv[g];

        // ---- X_t @ W contribution: K = 128 -> 4 bf16 k-steps ----
        const float*  xrowf = xrow0f + (size_t)t * D_;
        const __bf16* xrowb = xrow0b + (size_t)t * D_;
#pragma unroll
        for (int kb = 0; kb < 4; ++kb) {
            // bf16 A 16x32 layout: lanes 0-15 K {0..7,16..23}, lanes 16-31 K {8..15,24..31}
            v16bf a;
            if constexpr (PRECONV) {
                v8bf a0 = *(const v8bf*)(xrowb + kb * 32 + hi * 8);
                v8bf a1 = *(const v8bf*)(xrowb + kb * 32 + 16 + hi * 8);
#pragma unroll
                for (int i = 0; i < 8; ++i) { a[i] = a0[i]; a[i + 8] = a1[i]; }
            } else {
                float tmp[16];
                const float4* p1 = (const float4*)(xrowf + kb * 32 + hi * 8);
                const float4* p2 = (const float4*)(xrowf + kb * 32 + 16 + hi * 8);
                *(float4*)(tmp + 0)  = p1[0]; *(float4*)(tmp + 4)  = p1[1];
                *(float4*)(tmp + 8)  = p2[0]; *(float4*)(tmp + 12) = p2[1];
#pragma unroll
                for (int i = 0; i < 16; ++i) a[i] = (__bf16)tmp[i];
            }
#pragma unroll
            for (int g = 0; g < 4; ++g) {
                // bf16 B 32x16 layout: lanes 0-15 K {0..7,16..23}, lanes 16-31 K {8..15,24..31}
                v8bf blo = *(const v8bf*)&ldsW[g][col][kb * 32 + hi * 8];
                v8bf bhi = *(const v8bf*)&ldsW[g][col][kb * 32 + 16 + hi * 8];
                v16bf bf;
#pragma unroll
                for (int i = 0; i < 8; ++i) { bf[i] = blo[i]; bf[i + 8] = bhi[i]; }
                acc[g] = __builtin_amdgcn_wmma_f32_16x16x32_bf16(
                    false, a, false, bf, (short)0, acc[g], false, false);
            }
        }

        // ---- h_{t-1} @ U contribution: K = 256 -> 8 bf16 k-steps ----
        const __bf16* hrow = hbuf + (size_t)(t & 1) * B_ * H_ + (size_t)arow * H_;
#pragma unroll
        for (int kb = 0; kb < 8; ++kb) {
            v8bf a0 = *(const v8bf*)(hrow + kb * 32 + hi * 8);
            v8bf a1 = *(const v8bf*)(hrow + kb * 32 + 16 + hi * 8);
            v16bf a;
#pragma unroll
            for (int i = 0; i < 8; ++i) { a[i] = a0[i]; a[i + 8] = a1[i]; }
#pragma unroll
            for (int g = 0; g < 4; ++g) {
                v8bf blo = *(const v8bf*)&ldsU[g][col][kb * 32 + hi * 8];
                v8bf bhi = *(const v8bf*)&ldsU[g][col][kb * 32 + 16 + hi * 8];
                v16bf bf;
#pragma unroll
                for (int i = 0; i < 8; ++i) { bf[i] = blo[i]; bf[i + 8] = bhi[i]; }
                acc[g] = __builtin_amdgcn_wmma_f32_16x16x32_bf16(
                    false, a, false, bf, (short)0, acc[g], false, false);
            }
        }

        // ---- gates + state update (wave-local, f32) ----
#pragma unroll
        for (int r = 0; r < 8; ++r) {
            float ig = sigmoidf_(acc[0][r]);
            float fg = sigmoidf_(acc[1][r]);
            float gg = tanhf(acc[2][r]);
            float og = sigmoidf_(acc[3][r]);
            float cn = fg * creg[r] + ig * gg;
            creg[r] = cn;
            hreg[r] = og * tanhf(cn);
        }

        // ---- write hidden_seq (f32) and next-step h buffer (bf16) ----
        __bf16* nb = hbuf + (size_t)((t + 1) & 1) * B_ * H_;
#pragma unroll
        for (int r = 0; r < 8; ++r) {
            int b = b0 + r + hi * 8;
            out[((size_t)b * T_ + t) * H_ + jc] = hreg[r];
            nb[(size_t)b * H_ + jc] = (__bf16)hreg[r];
        }

        // Pull next timestep's X row toward the caches while we sync.
        if (t + 1 < T_) {
            if constexpr (PRECONV) __builtin_prefetch(xrowb + D_, 0, 1);
            else                   __builtin_prefetch(xrowf + D_, 0, 1);
        }

        // ---- grid-wide barrier (release h writes, acquire for next step) ----
        __threadfence();
        __syncthreads();
        if (tid == 0) {
            __hip_atomic_fetch_add(bar, 1u, __ATOMIC_RELEASE, __HIP_MEMORY_SCOPE_AGENT);
            unsigned target = (unsigned)(t + 1) * NWG;
            while (__hip_atomic_load(bar, __ATOMIC_ACQUIRE, __HIP_MEMORY_SCOPE_AGENT) < target)
                __builtin_amdgcn_s_sleep(1);
        }
        __syncthreads();
        __threadfence();
    }

    // ---- final h_t, c_t ----
    const size_t off_h = (size_t)B_ * T_ * H_;
    const size_t off_c = off_h + (size_t)B_ * H_;
#pragma unroll
    for (int r = 0; r < 8; ++r) {
        int b = b0 + r + hi * 8;
        out[off_h + (size_t)b * H_ + jc] = hreg[r];
        out[off_c + (size_t)b * H_ + jc] = creg[r];
    }
}

extern "C" void kernel_launch(void* const* d_in, const int* in_sizes, int n_in,
                              void* d_out, int out_size, void* d_ws, size_t ws_size,
                              hipStream_t stream) {
    const float* X    = (const float*)d_in[0];
    const float* W    = (const float*)d_in[1];
    const float* U    = (const float*)d_in[2];
    const float* bias = (const float*)d_in[3];
    const float* h0   = (const float*)d_in[4];
    const float* c0   = (const float*)d_in[5];
    float* out = (float*)d_out;

    unsigned* bar  = (unsigned*)d_ws;                    // 4B counter
    __bf16*   hbuf = (__bf16*)((char*)d_ws + 256);       // 2 x [64x256] bf16

    lstm_init_kernel<<<(B_ * H_ + 255) / 256, 256, 0, stream>>>(h0, hbuf, bar);

    const bool preconv = ws_size >= (size_t)WS_XBF_OFF + WS_XBF_SZ;
    __bf16* Xbf = (__bf16*)((char*)d_ws + WS_XBF_OFF);

    if (preconv) {
        const size_t n8 = (size_t)B_ * T_ * D_ / 8;
        x_to_bf16_kernel<<<(unsigned)((n8 + 255) / 256), 256, 0, stream>>>(X, Xbf);
        lstm_persistent_kernel<true><<<NWG, 128, 0, stream>>>(
            X, Xbf, W, U, bias, c0, out, hbuf, bar);
    } else {
        lstm_persistent_kernel<false><<<NWG, 128, 0, stream>>>(
            X, Xbf, W, U, bias, c0, out, hbuf, bar);
    }
}
